// GNNAgent_87857851007435
// MI455X (gfx1250) — matrix operand order repaired and checked by reading, not compiled, not run on hardware
//
#include <hip/hip_runtime.h>
#include <hip/hip_bf16.h>

#define N_AGENTS  32
#define HIDDEN    256
#define OBS       128
#define N_ACTIONS 16
#define NUM_LAYERS 4
#define N_ENVS    4096
#define N_ROWS    (N_ENVS * N_AGENTS)

typedef __attribute__((ext_vector_type(16))) __bf16 v16bf;
typedef __attribute__((ext_vector_type(8)))  __bf16 v8bf;
typedef __attribute__((ext_vector_type(8)))  float  v8f;

// Workspace offsets (in bf16 elements): swizzled weight tiles.
#define OFS_ENC 0
#define OFS_M1  (OFS_ENC + OBS * HIDDEN)        // 32768
#define OFS_M2  (OFS_M1  + HIDDEN * HIDDEN)     // 98304
#define OFS_POL (OFS_M2  + HIDDEN * HIDDEN)     // 163840
#define OFS_ACT (OFS_POL + 2 * HIDDEN * HIDDEN) // 294912

// ---------------------------------------------------------------------------
// Weight pre-swizzle: f32 row-major (Ktot x Ntot) -> bf16 WMMA-B fragment
// layout. Tile (kt,nt) of 32(K)x16(N); within a tile element (lane,i) maps to
//   k = i + 16*(lane/16) + 32*kt ,  n = (lane%16) + 16*nt
// so every lane's 16 bf16 values are contiguous (32B) in memory.
// ---------------------------------------------------------------------------
__global__ void swizzle_w(const float* __restrict__ src, __bf16* __restrict__ dst,
                          int Ktot, int Ntot) {
  int idx = blockIdx.x * 256 + threadIdx.x;
  if (idx >= Ktot * Ntot) return;
  int k = idx / Ntot, n = idx % Ntot;
  int kt = k >> 5, r = k & 31, half = r >> 4, i = r & 15;
  int nt = n >> 4, lane = (n & 15) + (half << 4);
  int ktiles = Ktot >> 5;
  dst[(((nt * ktiles + kt) << 9) + (lane << 4)) + i] = (__bf16)src[idx];
}

// ---------------------------------------------------------------------------
// Fragment helpers (wave32, CDNA5 16x16x32 bf16 layouts)
// ---------------------------------------------------------------------------
// A (16x32): lane holds row m=lane%16 (+16*rt); K = i + 16*half
// => two contiguous 16B LDS loads per fragment.
__device__ __forceinline__ v16bf load_a_lds(const __bf16* base, int stride,
                                            int rt, int kbase, int lane) {
  const int m = (lane & 15) + (rt << 4);
  const int half = lane >> 4;
  const __bf16* p = base + m * stride + kbase + 16 * half;
  v8bf lo = *(const v8bf*)p;
  v8bf hi = *(const v8bf*)(p + 8);
  v16bf a;
#pragma unroll
  for (int i = 0; i < 8; ++i) { a[i] = lo[i]; a[i + 8] = hi[i]; }
  return a;
}

// B from pre-swizzled global weights: one contiguous 32B load per lane.
__device__ __forceinline__ v16bf load_b_w(const __bf16* wbase, int ktiles,
                                          int nt, int kt, int lane) {
  return *(const v16bf*)(wbase + ((size_t)(nt * ktiles + kt) << 9) + (lane << 4));
}

// B (32x16) from row-major LDS activation (for adj @ msg): lane holds column
// n=lane%16 (+16*nt), K = i + 16*half -> strided u16 gathers.
__device__ __forceinline__ v16bf load_b_lds_rm(const __bf16* base, int stride,
                                               int nt, int lane) {
  const int n = (lane & 15) + (nt << 4);
  const int half = lane >> 4;
  v16bf b;
#pragma unroll
  for (int i = 0; i < 16; ++i) b[i] = base[(i + 16 * half) * stride + n];
  return b;
}

__device__ __forceinline__ v8f wmma_bf16(v16bf a, v16bf b, v8f c) {
  // v_wmma_f32_16x16x32_bf16  (f32 accumulate)
  return __builtin_amdgcn_wmma_f32_16x16x32_bf16(false, a, false, b,
                                                 (short)0, c, false, false);
}

// C/D: lane holds col n=lane%16; VGPR v -> row m = v + 8*half (+16*rt).
__device__ __forceinline__ void store_c_bf16(__bf16* base, int stride, int rt,
                                             int nt, v8f c, int lane, bool relu) {
  const int n = (lane & 15) + (nt << 4);
  const int half = lane >> 4;
#pragma unroll
  for (int v = 0; v < 8; ++v) {
    float f = c[v];
    if (relu) f = fmaxf(f, 0.0f);
    base[(v + 8 * half + 16 * rt) * stride + n] = (__bf16)f;
  }
}

// Generic phase: Out(32xN) = act(A(32xK) @ Wswz) [+bias][relu].
// ks-outer / 4-accumulator form: each B fragment loaded once, used by both
// row-tiles; each A fragment used by both column-tiles. ~100 live VGPRs.
__device__ __forceinline__ void gemm_phase(const __bf16* A, int a_stride, int ksteps,
                                           const __bf16* W, __bf16* Out,
                                           const float* bias, bool relu,
                                           int lane, int wave) {
  const int nt0 = wave * 2, nt1 = nt0 + 1;
  // Pull this wave's weight panel toward L2/L0 early (global_prefetch_b8).
  __builtin_prefetch(W + ((size_t)(nt0 * ksteps) << 9), 0, 1);
  v8f c00 = {}, c01 = {}, c10 = {}, c11 = {};
#pragma unroll 2
  for (int ks = 0; ks < ksteps; ++ks) {
    v16bf b0 = load_b_w(W, ksteps, nt0, ks, lane);
    v16bf b1 = load_b_w(W, ksteps, nt1, ks, lane);
    v16bf a0 = load_a_lds(A, a_stride, 0, ks * 32, lane);
    v16bf a1 = load_a_lds(A, a_stride, 1, ks * 32, lane);
    c00 = wmma_bf16(a0, b0, c00);
    c01 = wmma_bf16(a0, b1, c01);
    c10 = wmma_bf16(a1, b0, c10);
    c11 = wmma_bf16(a1, b1, c11);
  }
  if (bias) {
    float f0 = bias[(lane & 15) + (nt0 << 4)];
    float f1 = bias[(lane & 15) + (nt1 << 4)];
#pragma unroll
    for (int v = 0; v < 8; ++v) {
      c00[v] += f0; c10[v] += f0;
      c01[v] += f1; c11[v] += f1;
    }
  }
  store_c_bf16(Out, HIDDEN, 0, nt0, c00, lane, relu);
  store_c_bf16(Out, HIDDEN, 0, nt1, c01, lane, relu);
  store_c_bf16(Out, HIDDEN, 1, nt0, c10, lane, relu);
  store_c_bf16(Out, HIDDEN, 1, nt1, c11, lane, relu);
}

// agg = adj(32x32) @ msg(32x256): single K-step, B sourced from LDS.
__device__ __forceinline__ void gemm_agg(const __bf16* adjm, const __bf16* msg,
                                         __bf16* Out, int lane, int wave) {
  const int nt0 = wave * 2, nt1 = nt0 + 1;
  v16bf b0 = load_b_lds_rm(msg, HIDDEN, nt0, lane);
  v16bf b1 = load_b_lds_rm(msg, HIDDEN, nt1, lane);
  v16bf a0 = load_a_lds(adjm, N_AGENTS, 0, 0, lane);
  v16bf a1 = load_a_lds(adjm, N_AGENTS, 1, 0, lane);
  v8f c00 = {}, c01 = {}, c10 = {}, c11 = {};
  c00 = wmma_bf16(a0, b0, c00);
  c01 = wmma_bf16(a0, b1, c01);
  c10 = wmma_bf16(a1, b0, c10);
  c11 = wmma_bf16(a1, b1, c11);
  store_c_bf16(Out, HIDDEN, 0, nt0, c00, lane, false);
  store_c_bf16(Out, HIDDEN, 0, nt1, c01, lane, false);
  store_c_bf16(Out, HIDDEN, 1, nt0, c10, lane, false);
  store_c_bf16(Out, HIDDEN, 1, nt1, c11, lane, false);
}

// ---------------------------------------------------------------------------
// Fully fused per-env kernel: 1 workgroup (8 wave32) per env.
// amdgpu_waves_per_eu(4): <=256 VGPRs/wave; the 4-accumulator GEMM form keeps
// live state ~100-160 VGPRs so this should hold without scratch spills.
// ---------------------------------------------------------------------------
__global__ __launch_bounds__(256)
__attribute__((amdgpu_waves_per_eu(4)))
void gnn_fused(
    const float* __restrict__ x, const float* __restrict__ adj,
    const __bf16* __restrict__ ws,
    const float* __restrict__ b_enc, const float* __restrict__ b_pol,
    const float* __restrict__ b_act,
    float* __restrict__ out_act, float* __restrict__ out_h) {
  __shared__ __bf16 ls_x[N_AGENTS * OBS];           //  8 KB
  __shared__ __bf16 ls_adj[N_AGENTS * N_AGENTS];    //  2 KB
  __shared__ __bf16 ls_enc[N_AGENTS * HIDDEN];      // 16 KB (kept for concat)
  __shared__ __bf16 ls_b0[N_AGENTS * HIDDEN];       // 16 KB ping
  __shared__ __bf16 ls_b1[N_AGENTS * HIDDEN];       // 16 KB pong

  const int tid = threadIdx.x;
  const int lane = tid & 31;
  const int wave = tid >> 5;
  const int env = blockIdx.x;
  const int rowbase = env * N_AGENTS;

  // Stage x (32x128 f32 -> bf16) and adj (32x32) into LDS, coalesced.
  for (int i = tid; i < N_AGENTS * OBS; i += 256)
    ls_x[i] = (__bf16)x[(size_t)rowbase * OBS + i];
  for (int i = tid; i < N_AGENTS * N_AGENTS; i += 256)
    ls_adj[i] = (__bf16)adj[(size_t)env * N_AGENTS * N_AGENTS + i];
  __syncthreads();

  // enc = relu(x @ W_enc + b_enc)            (K = 128 -> 4 k-steps)
  gemm_phase(ls_x, OBS, OBS / 32, ws + OFS_ENC, ls_enc, b_enc, true, lane, wave);
  __syncthreads();

  // 4 message-passing layers, ping-pong msg between ls_b0/ls_b1.
  const __bf16* msrc = ls_enc;
  __bf16* pa = ls_b0;
  __bf16* pb = ls_b1;
  for (int layer = 0; layer < NUM_LAYERS; ++layer) {
    gemm_agg(ls_adj, msrc, pa, lane, wave);                                // agg
    __syncthreads();
    gemm_phase(pa, HIDDEN, 8, ws + OFS_M1, pb, nullptr, true, lane, wave);  // relu(agg@W1)
    __syncthreads();
    gemm_phase(pb, HIDDEN, 8, ws + OFS_M2, pa, nullptr, false, lane, wave); // @W2
    __syncthreads();
    msrc = pa;
    __bf16* t = pa; pa = pb; pb = t;
  }
  __bf16* hbuf = pa;  // free buffer (distinct from msrc)

  // h = relu([enc | msg] @ W_pol + b_pol)    (K = 512 -> 16 k-steps)
  {
    const int nt0 = wave * 2, nt1 = nt0 + 1;
    const __bf16* Wp = ws + OFS_POL;
    __builtin_prefetch(Wp + ((size_t)(nt0 * 16) << 9), 0, 1);
    v8f c00 = {}, c01 = {}, c10 = {}, c11 = {};
#pragma unroll 2
    for (int ks = 0; ks < 16; ++ks) {
      const __bf16* Asrc = (ks < 8) ? ls_enc : msrc;
      const int kb = (ks & 7) * 32;
      v16bf b0 = load_b_w(Wp, 16, nt0, ks, lane);
      v16bf b1 = load_b_w(Wp, 16, nt1, ks, lane);
      v16bf a0 = load_a_lds(Asrc, HIDDEN, 0, kb, lane);
      v16bf a1 = load_a_lds(Asrc, HIDDEN, 1, kb, lane);
      c00 = wmma_bf16(a0, b0, c00);
      c01 = wmma_bf16(a0, b1, c01);
      c10 = wmma_bf16(a1, b0, c10);
      c11 = wmma_bf16(a1, b1, c11);
    }
    const int half = lane >> 4;
    const int n0 = (lane & 15) + (nt0 << 4);
    const int n1 = (lane & 15) + (nt1 << 4);
    const float f0 = b_pol[n0], f1 = b_pol[n1];
#pragma unroll
    for (int v = 0; v < 8; ++v) {
      const int m0 = v + 8 * half;        // rt = 0
      const int m1 = m0 + 16;             // rt = 1
      float h00 = fmaxf(c00[v] + f0, 0.0f);
      float h01 = fmaxf(c01[v] + f1, 0.0f);
      float h10 = fmaxf(c10[v] + f0, 0.0f);
      float h11 = fmaxf(c11[v] + f1, 0.0f);
      hbuf[m0 * HIDDEN + n0] = (__bf16)h00;
      hbuf[m0 * HIDDEN + n1] = (__bf16)h01;
      hbuf[m1 * HIDDEN + n0] = (__bf16)h10;
      hbuf[m1 * HIDDEN + n1] = (__bf16)h11;
      out_h[(size_t)(rowbase + m0) * HIDDEN + n0] = h00;
      out_h[(size_t)(rowbase + m0) * HIDDEN + n1] = h01;
      out_h[(size_t)(rowbase + m1) * HIDDEN + n0] = h10;
      out_h[(size_t)(rowbase + m1) * HIDDEN + n1] = h11;
    }
  }
  __syncthreads();

  // actions = h @ W_act + b_act              (32x16 output: waves 0,1)
  if (wave < 2) {
    const int rt = wave;
    const __bf16* Wa = ws + OFS_ACT;
    v8f c = {};
#pragma unroll 2
    for (int ks = 0; ks < 8; ++ks) {
      v16bf a = load_a_lds(hbuf, HIDDEN, rt, ks * 32, lane);
      c = wmma_bf16(a, load_b_w(Wa, 8, 0, ks, lane), c);
    }
    const int half = lane >> 4;
    const int n = lane & 15;
    const float fb = b_act[n];
#pragma unroll
    for (int v = 0; v < 8; ++v) {
      const int m = v + 8 * half + 16 * rt;
      out_act[(size_t)(rowbase + m) * N_ACTIONS + n] = c[v] + fb;
    }
  }
}

// ---------------------------------------------------------------------------
extern "C" void kernel_launch(void* const* d_in, const int* in_sizes, int n_in,
                              void* d_out, int out_size, void* d_ws, size_t ws_size,
                              hipStream_t stream) {
  const float* x     = (const float*)d_in[0];
  const float* adj   = (const float*)d_in[1];
  const float* W_enc = (const float*)d_in[2];
  const float* b_enc = (const float*)d_in[3];
  const float* W_m1  = (const float*)d_in[4];
  const float* W_m2  = (const float*)d_in[5];
  const float* W_pol = (const float*)d_in[6];
  const float* b_pol = (const float*)d_in[7];
  const float* W_act = (const float*)d_in[8];
  const float* b_act = (const float*)d_in[9];

  float* out      = (float*)d_out;
  float* out_act  = out;                                 // (131072 x 16)
  float* out_h    = out + (size_t)N_ROWS * N_ACTIONS;    // (131072 x 256)
  __bf16* ws      = (__bf16*)d_ws;                       // ~584 KB used

  // Pre-swizzle weights to bf16 WMMA-B tile layout (re-run every call: deterministic).
  {
    int t;
    t = OBS * HIDDEN;
    swizzle_w<<<(t + 255) / 256, 256, 0, stream>>>(W_enc, ws + OFS_ENC, OBS, HIDDEN);
    t = HIDDEN * HIDDEN;
    swizzle_w<<<(t + 255) / 256, 256, 0, stream>>>(W_m1, ws + OFS_M1, HIDDEN, HIDDEN);
    swizzle_w<<<(t + 255) / 256, 256, 0, stream>>>(W_m2, ws + OFS_M2, HIDDEN, HIDDEN);
    t = 2 * HIDDEN * HIDDEN;
    swizzle_w<<<(t + 255) / 256, 256, 0, stream>>>(W_pol, ws + OFS_POL, 2 * HIDDEN, HIDDEN);
    t = HIDDEN * N_ACTIONS;
    swizzle_w<<<(t + 255) / 256, 256, 0, stream>>>(W_act, ws + OFS_ACT, HIDDEN, N_ACTIONS);
  }

  gnn_fused<<<N_ENVS, 256, 0, stream>>>(x, adj, ws, b_enc, b_pol, b_act,
                                        out_act, out_h);
}
